// BehaviorModel_1494648619261
// MI455X (gfx1250) — matrix operand order repaired and checked by reading, not compiled, not run on hardware
//
#include <hip/hip_runtime.h>
#include <hip/hip_bf16.h>

// ---------------- model constants ----------------
#define BB   32
#define TT   128
#define CCH  8
#define NND  128
#define HIDN 512
#define NE   2048
#define NE2  2176   // NE + NND self loops
#define FEAT 1024   // HEADS * fo = 2*512
#define GDIM 2048   // 4*HID lstm gates

typedef __attribute__((ext_vector_type(16))) _Float16 v16h;
typedef __attribute__((ext_vector_type(8)))  _Float16 v8h;
typedef __attribute__((ext_vector_type(8)))  float    v8f;

__device__ inline float lrelu(float v){ return v > 0.f ? v : 0.1f * v; }
__device__ inline float sigm(float v){ return 1.f / (1.f + expf(-v)); }
__device__ inline int   imin(int a, int b){ return a < b ? a : b; }

__device__ inline void edge_sd(const int* __restrict__ ei, int e, int& s, int& d){
  if (e < NE) { s = ei[e]; d = ei[NE + e]; }
  else        { s = e - NE; d = e - NE; }
}

__device__ inline float waveReduceSum(float v){
#pragma unroll
  for (int off = 16; off > 0; off >>= 1) v += __shfl_xor(v, off, 32);
  return v;
}

// ---------------- generic WMMA GEMM ----------------
// C[M,N] = act(A[M,K] @ B + bias). BT=0: B is KxN row-major. BT=1: B is NxK
// row-major (multiply by stored B transposed). Batched over blockIdx.z.
// Block = 128 threads = 4 waves, block tile 32x64, each wave: 16x32 (2 WMMAs).
// GUARD=false requires M%32==0, N%64==0, K%32==0, 16B-aligned rows.
// GUARD=true is branch-free: clamped addresses + value select (EXEC stays full).
// act: 0=none 1=leaky_relu(0.1) 2=sigmoid
template<bool BT, bool GUARD>
__global__ __launch_bounds__(128)
void gemm_wmma(const float* __restrict__ A, const float* __restrict__ Bm,
               const float* __restrict__ bias, float* __restrict__ C,
               int M, int N, int K, int lda, int ldb, int ldc,
               long sA, long sB, long sC, int act)
{
  __shared__ __attribute__((aligned(16))) _Float16 bt[64][40]; // transposed B tile (+pad)

  const int tid  = threadIdx.x;
  const int lane = tid & 31;
  const int wave = tid >> 5;
  const int kh   = lane >> 4;        // half-wave
  const int l16  = lane & 15;
  const int tM   = blockIdx.x * 32 + (wave >> 1) * 16;
  const int tN0  = blockIdx.y * 64;                 // block N origin
  const int nl0  = (wave & 1) * 32 + l16;           // lane's first local col
  const int col0 = tN0 + nl0;                       // global col, sub-tile 0
  const int mrow = tM + l16;

  A  += (long)blockIdx.z * sA;
  Bm += (long)blockIdx.z * sB;
  C  += (long)blockIdx.z * sC;

  v8f acc[2];
  acc[0] = (v8f){0.f,0.f,0.f,0.f,0.f,0.f,0.f,0.f};
  acc[1] = acc[0];

  const float* pa = A + (long)(GUARD ? imin(mrow, M - 1) : mrow) * lda;
  const bool mok = mrow < M;

  for (int k0 = 0; k0 < K; k0 += 32) {
    // ---- A fragment: 16x32 f16: lanes 0-15 K 0..15, hi half K 16..31
    v16h af;
    if (GUARD) {
#pragma unroll
      for (int v = 0; v < 8; ++v) {
        int kb = k0 + kh * 16 + v * 2;
        float f0 = pa[imin(kb,     K - 1)];
        float f1 = pa[imin(kb + 1, K - 1)];
        af[2*v]   = (_Float16)((mok && kb     < K) ? f0 : 0.f);
        af[2*v+1] = (_Float16)((mok && kb + 1 < K) ? f1 : 0.f);
      }
    } else {
      __builtin_prefetch(pa + k0 + 64, 0, 0);   // global_prefetch_b8 next tiles
      const float4 a0 = *(const float4*)(pa + k0 + kh * 16);
      const float4 a1 = *(const float4*)(pa + k0 + kh * 16 + 4);
      const float4 a2 = *(const float4*)(pa + k0 + kh * 16 + 8);
      const float4 a3 = *(const float4*)(pa + k0 + kh * 16 + 12);
      af[0]=(_Float16)a0.x; af[1]=(_Float16)a0.y; af[2]=(_Float16)a0.z; af[3]=(_Float16)a0.w;
      af[4]=(_Float16)a1.x; af[5]=(_Float16)a1.y; af[6]=(_Float16)a1.z; af[7]=(_Float16)a1.w;
      af[8]=(_Float16)a2.x; af[9]=(_Float16)a2.y; af[10]=(_Float16)a2.z; af[11]=(_Float16)a2.w;
      af[12]=(_Float16)a3.x;af[13]=(_Float16)a3.y;af[14]=(_Float16)a3.z; af[15]=(_Float16)a3.w;
    }

    // ---- B fragments (2 sub-tiles): 32x16 f16; lanes 0-15 K 0..15, hi half K 16..31
    v16h bf[2];
    if (GUARD) {
#pragma unroll
      for (int s = 0; s < 2; ++s) {
        int col  = col0 + s * 16;
        int colc = imin(col, N - 1);
        bool cok = col < N;
#pragma unroll
        for (int e = 0; e < 16; ++e) {
          int kk = k0 + kh * 16 + e;
          int kc = imin(kk, K - 1);
          float f = BT ? Bm[(long)colc * ldb + kc] : Bm[(long)kc * ldb + colc];
          bf[s][e] = (_Float16)((cok && kk < K) ? f : 0.f);
        }
      }
    } else if (BT) {
#pragma unroll
      for (int s = 0; s < 2; ++s) {
        const float* pb = Bm + (long)(col0 + s * 16) * ldb + k0 + kh * 16;
        const float4 b0 = ((const float4*)pb)[0];
        const float4 b1 = ((const float4*)pb)[1];
        const float4 b2 = ((const float4*)pb)[2];
        const float4 b3 = ((const float4*)pb)[3];
        bf[s][0]=(_Float16)b0.x; bf[s][1]=(_Float16)b0.y; bf[s][2]=(_Float16)b0.z; bf[s][3]=(_Float16)b0.w;
        bf[s][4]=(_Float16)b1.x; bf[s][5]=(_Float16)b1.y; bf[s][6]=(_Float16)b1.z; bf[s][7]=(_Float16)b1.w;
        bf[s][8]=(_Float16)b2.x; bf[s][9]=(_Float16)b2.y; bf[s][10]=(_Float16)b2.z; bf[s][11]=(_Float16)b2.w;
        bf[s][12]=(_Float16)b3.x;bf[s][13]=(_Float16)b3.y;bf[s][14]=(_Float16)b3.z; bf[s][15]=(_Float16)b3.w;
      }
    } else {
      // cooperative staging of 32x64 B tile, transposed into LDS
      __syncthreads();
      {
        const int kk = tid >> 3;            // 0..15
        const int nn = (tid & 7) * 8;       // 0..56
        const float* r0 = Bm + (long)(k0 + kk) * ldb + tN0 + nn;
        const float* r1 = Bm + (long)(k0 + kk + 16) * ldb + tN0 + nn;
        const float4 v0 = ((const float4*)r0)[0];
        const float4 v1 = ((const float4*)r0)[1];
        const float4 v2 = ((const float4*)r1)[0];
        const float4 v3 = ((const float4*)r1)[1];
        bt[nn+0][kk]    = (_Float16)v0.x; bt[nn+1][kk]    = (_Float16)v0.y;
        bt[nn+2][kk]    = (_Float16)v0.z; bt[nn+3][kk]    = (_Float16)v0.w;
        bt[nn+4][kk]    = (_Float16)v1.x; bt[nn+5][kk]    = (_Float16)v1.y;
        bt[nn+6][kk]    = (_Float16)v1.z; bt[nn+7][kk]    = (_Float16)v1.w;
        bt[nn+0][kk+16] = (_Float16)v2.x; bt[nn+1][kk+16] = (_Float16)v2.y;
        bt[nn+2][kk+16] = (_Float16)v2.z; bt[nn+3][kk+16] = (_Float16)v2.w;
        bt[nn+4][kk+16] = (_Float16)v3.x; bt[nn+5][kk+16] = (_Float16)v3.y;
        bt[nn+6][kk+16] = (_Float16)v3.z; bt[nn+7][kk+16] = (_Float16)v3.w;
      }
      __syncthreads();
      bf[0] = *(const v16h*)&bt[nl0][kh * 16];
      bf[1] = *(const v16h*)&bt[nl0 + 16][kh * 16];
    }

#pragma unroll
    for (int s = 0; s < 2; ++s)
      acc[s] = __builtin_amdgcn_wmma_f32_16x16x32_f16(false, af, false, bf[s],
                                                      (short)0, acc[s], false, false);
  }

  // ---- epilogue: C/D layout: VGPR r -> M = r + 8*kh, N = lane&15
#pragma unroll
  for (int s = 0; s < 2; ++s) {
    int col = col0 + s * 16;
#pragma unroll
    for (int r = 0; r < 8; ++r) {
      int m = tM + r + 8 * kh;
      if (!GUARD || (m < M && col < N)) {
        float v = acc[s][r];
        if (bias) v += bias[col];
        if (act == 1)      v = lrelu(v);
        else if (act == 2) v = sigm(v);
        C[(long)m * ldc + col] = v;
      }
    }
  }
}

// ---------------- fused LSTM recurrence step ----------------
// One block per 16-wide hidden slice (grid.x = HIDN/16 = 32).
// Block covers all 32 batch rows and all 4 gate chunks:
//   wave w: rows (w>>1)*16..+15, gate chunks {(w&1)*2, (w&1)*2+1} (2 WMMAs).
// h kept in f16 (ping-pong), Whh pre-converted to f16 -> no cvt in inner loop.
__global__ __launch_bounds__(128)
void k_lstm_step(const _Float16* __restrict__ hin,   // 32 x 512 f16
                 const _Float16* __restrict__ whh,   // 2048 x 512 f16 (NxK)
                 const float* __restrict__ pre,      // (B*T, 2048) f32
                 float* __restrict__ cs,             // 32 x 512 f32
                 _Float16* __restrict__ hout,        // 32 x 512 f16
                 float* __restrict__ lo,             // (B,T,1024) f32
                 int t, int dir)
{
  __shared__ __attribute__((aligned(16))) float gacc[4][32][16];

  const int tid  = threadIdx.x;
  const int lane = tid & 31;
  const int wave = tid >> 5;
  const int kh   = lane >> 4;
  const int l16  = lane & 15;
  const int rbase = (wave >> 1) * 16;
  const int j0 = blockIdx.x * 16;

  v8f acc[2];
  acc[0] = (v8f){0.f,0.f,0.f,0.f,0.f,0.f,0.f,0.f};
  acc[1] = acc[0];

  const _Float16* pa = hin + (long)(rbase + l16) * HIDN;
  for (int k0 = 0; k0 < HIDN; k0 += 32) {
    v16h af;
    const v8h alo = *(const v8h*)(pa + k0 + kh * 16);
    const v8h ahi = *(const v8h*)(pa + k0 + kh * 16 + 8);
#pragma unroll
    for (int i = 0; i < 8; ++i) { af[i] = alo[i]; af[8 + i] = ahi[i]; }
#pragma unroll
    for (int s = 0; s < 2; ++s) {
      const int g = (wave & 1) * 2 + s;
      const v16h bfr = *(const v16h*)(whh + (long)(g * HIDN + j0 + l16) * HIDN
                                          + k0 + kh * 16);
      acc[s] = __builtin_amdgcn_wmma_f32_16x16x32_f16(false, af, false, bfr,
                                                      (short)0, acc[s], false, false);
    }
  }

  // spill gate tiles to LDS: gacc[gate][batch_row][col]
#pragma unroll
  for (int s = 0; s < 2; ++s) {
    const int g = (wave & 1) * 2 + s;
#pragma unroll
    for (int r = 0; r < 8; ++r)
      gacc[g][rbase + r + 8 * kh][l16] = acc[s][r];
  }
  __syncthreads();

  // gate math: 512 elements (32 rows x 16 cols), 4 per thread
#pragma unroll
  for (int q = 0; q < 4; ++q) {
    int e = tid + 128 * q;
    int row = e >> 4, col = e & 15;
    int j = j0 + col;
    long pb = ((long)row * TT + t) * GDIM;
    float gi = pre[pb + 0 * HIDN + j] + gacc[0][row][col];
    float gf = pre[pb + 1 * HIDN + j] + gacc[1][row][col];
    float gg = pre[pb + 2 * HIDN + j] + gacc[2][row][col];
    float go = pre[pb + 3 * HIDN + j] + gacc[3][row][col];
    int ci = row * HIDN + j;
    float cv = sigm(gf) * cs[ci] + sigm(gi) * tanhf(gg);
    float hv = sigm(go) * tanhf(cv);
    cs[ci] = cv;
    hout[ci] = (_Float16)hv;
    lo[((long)row * TT + t) * (2 * HIDN) + dir * HIDN + j] = hv;
  }
}

// ---------------- small utility kernels ----------------
__global__ void k_fill(float* p, long n, float v){
  long i = (long)blockIdx.x * blockDim.x + threadIdx.x;
  if (i < n) p[i] = v;
}

__global__ void k_f2h(const float* __restrict__ in, _Float16* __restrict__ o, long n){
  long i = (long)blockIdx.x * blockDim.x + threadIdx.x;
  if (i < n) o[i] = (_Float16)in[i];
}

__global__ void k_add2(const float* a, const float* b, float* c, int n){
  int i = blockIdx.x * blockDim.x + threadIdx.x;
  if (i < n) c[i] = a[i] + b[i];
}

// h0[b,n,c] = x[b, T-1, c, n]
__global__ void k_h0(const float* __restrict__ x, float* __restrict__ h0){
  int i = blockIdx.x * blockDim.x + threadIdx.x;
  if (i >= BB * NND * CCH) return;
  int c = i & 7, n = (i >> 3) & 127, b = i >> 10;
  h0[i] = x[(((long)b * TT + (TT - 1)) * CCH + c) * NND + n];
}

// ---- CSR build (deterministic; graph shared by all batches/layers) ----
__global__ void k_fill_i(int* p, int n, int v){
  int i = blockIdx.x * blockDim.x + threadIdx.x;
  if (i < n) p[i] = v;
}

__global__ void k_deg(const int* __restrict__ ei, int* __restrict__ deg){
  int e = blockIdx.x * blockDim.x + threadIdx.x;
  if (e >= NE2) return;
  int s, d; edge_sd(ei, e, s, d); (void)s;
  atomicAdd(&deg[d], 1);
}

__global__ void k_scan(const int* __restrict__ deg, int* __restrict__ rowp){
  if (threadIdx.x == 0 && blockIdx.x == 0) {
    int acc = 0;
    for (int i = 0; i < NND; i++) { rowp[i] = acc; acc += deg[i]; }
    rowp[NND] = acc;
  }
}

// one thread per node scans all edges in ascending order -> deterministic CSR
__global__ void k_fillcsr(const int* __restrict__ ei, const int* __restrict__ rowp,
                          int* __restrict__ csre){
  int n = threadIdx.x;  // 128 threads
  int pos = rowp[n];
  for (int e = 0; e < NE2; e++) {
    int s, d; edge_sd(ei, e, s, d); (void)s;
    if (d == n) csre[pos++] = e;
  }
}

// wave per (e, b, h): logit = sum_c lrelu(xl[src]+xr[dst]) * att
__global__ void k_edge_logit(const float* __restrict__ xl, const float* __restrict__ xr,
                             const float* __restrict__ att, const int* __restrict__ ei,
                             float* __restrict__ logit){
  int lane = threadIdx.x & 31;
  int e = blockIdx.x * 4 + (threadIdx.x >> 5);
  int b = blockIdx.y, h = blockIdx.z;
  if (e >= NE2) return;
  int s, d; edge_sd(ei, e, s, d);
  const float* pl = xl + (long)(b * NND + s) * FEAT + h * HIDN;
  const float* pr = xr + (long)(b * NND + d) * FEAT + h * HIDN;
  const float* pa = att + h * HIDN;
  float acc = 0.f;
  for (int c = lane; c < HIDN; c += 32)
    acc += lrelu(pl[c] + pr[c]) * pa[c];
  acc = waveReduceSum(acc);
  if (lane == 0) logit[((long)b * NE2 + e) * 2 + h] = acc;
}

// per (b,n,h): segment max + softmax denominator over CSR in-edges (deterministic)
__global__ void k_node_maxden(const float* __restrict__ lg, const int* __restrict__ rowp,
                              const int* __restrict__ csre,
                              float* __restrict__ m, float* __restrict__ den){
  int n = blockIdx.x, b = blockIdx.y;
  int lane = threadIdx.x & 31, h = threadIdx.x >> 5;  // 64 threads = 2 waves
  int beg = rowp[n], end = rowp[n + 1];
  float mx = -3.0e38f;
  for (int q = beg + lane; q < end; q += 32)
    mx = fmaxf(mx, lg[((long)b * NE2 + csre[q]) * 2 + h]);
#pragma unroll
  for (int o = 16; o > 0; o >>= 1) mx = fmaxf(mx, __shfl_xor(mx, o, 32));
  float s = 0.f;
  for (int q = beg + lane; q < end; q += 32)
    s += expf(lg[((long)b * NE2 + csre[q]) * 2 + h] - mx);
  s = waveReduceSum(s);
  if (lane == 0) {
    m  [(b * NND + n) * 2 + h] = mx;
    den[(b * NND + n) * 2 + h] = s;
  }
}

// ex = exp(logit - m[dst])  (elementwise, in place, no atomics)
__global__ void k_edge_exp(float* __restrict__ logit, const float* __restrict__ mbuf,
                           const int* __restrict__ ei){
  int i = blockIdx.x * blockDim.x + threadIdx.x;
  if (i >= BB * NE2 * 2) return;
  int h = i & 1, e = (i >> 1) % NE2, b = i / (NE2 * 2);
  int s, d; edge_sd(ei, e, s, d); (void)s;
  logit[i] = expf(logit[i] - mbuf[(b * NND + d) * 2 + h]);
}

// dst-centric gather: out[b,n,f] = bias[f] + sum_{e in in(n)} alpha[e,h]*xl[b,src,f]
__global__ void k_gat_out(const float* __restrict__ ex, const float* __restrict__ den,
                          const float* __restrict__ xl, const int* __restrict__ ei,
                          const int* __restrict__ rowp, const int* __restrict__ csre,
                          const float* __restrict__ bias, float* __restrict__ o){
  int n = blockIdx.x, b = blockIdx.y;
  int tid = threadIdx.x;                    // 256 threads, 4 feats each
  int beg = rowp[n], end = rowp[n + 1];
  float d0 = den[(b * NND + n) * 2 + 0];
  float d1 = den[(b * NND + n) * 2 + 1];
  float acc[4];
#pragma unroll
  for (int j = 0; j < 4; j++) acc[j] = bias[tid + 256 * j];
  for (int q = beg; q < end; ++q) {
    int e = csre[q];
    int s, d; edge_sd(ei, e, s, d); (void)d;
    float a0 = ex[((long)b * NE2 + e) * 2 + 0] / d0;
    float a1 = ex[((long)b * NE2 + e) * 2 + 1] / d1;
    const float* px = xl + (long)(b * NND + s) * FEAT;
#pragma unroll
    for (int j = 0; j < 4; j++) {
      int f = tid + 256 * j;
      acc[j] += ((f < HIDN) ? a0 : a1) * px[f];
    }
  }
  float* po = o + (long)(b * NND + n) * FEAT;
#pragma unroll
  for (int j = 0; j < 4; j++) po[tid + 256 * j] = acc[j];
}

// wave per (t,b): wl[b,t] = sum_f out[b,t,f]*attnW[f] + attnB
__global__ void k_attn_logit(const float* __restrict__ o, const float* __restrict__ aw,
                             const float* __restrict__ ab, float* __restrict__ wl){
  int lane = threadIdx.x & 31;
  int t = blockIdx.x * 4 + (threadIdx.x >> 5);
  int b = blockIdx.y;
  const float* po = o + ((long)b * TT + t) * (2 * HIDN);
  float acc = 0.f;
  for (int f = lane; f < 2 * HIDN; f += 32) acc += po[f] * aw[f];
  acc = waveReduceSum(acc);
  if (lane == 0) wl[b * TT + t] = acc + ab[0];
}

// softmax over T per batch (block per b, 128 threads)
__global__ void k_softmax_T(float* __restrict__ w){
  __shared__ float sh[TT];
  __shared__ float sred;
  int b = blockIdx.x, t = threadIdx.x;
  float v = w[b * TT + t];
  sh[t] = v; __syncthreads();
  if (t == 0) { float m = sh[0]; for (int i = 1; i < TT; i++) m = fmaxf(m, sh[i]); sred = m; }
  __syncthreads();
  float ex = expf(v - sred);
  sh[t] = ex; __syncthreads();
  if (t == 0) { float s = 0.f; for (int i = 0; i < TT; i++) s += sh[i]; sred = s; }
  __syncthreads();
  w[b * TT + t] = ex / sred;
}

// la[b,f] = lrelu(sum_t w[b,t]*out[b,t,f])
__global__ void k_attn_pool(const float* __restrict__ w, const float* __restrict__ o,
                            float* __restrict__ la){
  int i = blockIdx.x * blockDim.x + threadIdx.x;
  if (i >= BB * 2 * HIDN) return;
  int b = i / (2 * HIDN), f = i % (2 * HIDN);
  float s = 0.f;
  for (int t = 0; t < TT; t++) s += w[b * TT + t] * o[((long)b * TT + t) * (2 * HIDN) + f];
  la[i] = lrelu(s);
}

// z[b, f*N + n] = h4[b,n,f]  (feature-major flatten of swapaxes);  z tail = la
__global__ void k_build_z(const float* __restrict__ h4, const float* __restrict__ la,
                          float* __restrict__ z){
  const long ZW = (long)FEAT * NND + 2 * HIDN;  // 132096
  long i = (long)blockIdx.x * blockDim.x + threadIdx.x;
  if (i >= BB * ZW) return;
  long b = i / ZW, r = i % ZW;
  float v;
  if (r < (long)FEAT * NND) {
    long f = r / NND, n = r % NND;
    v = h4[((long)b * NND + n) * FEAT + f];
  } else {
    v = la[b * (2 * HIDN) + (r - (long)FEAT * NND)];
  }
  z[i] = v;
}

// ---------------- host side ----------------
static inline dim3 g1(long n, int b){ return dim3((unsigned)((n + b - 1) / b)); }

extern "C" void kernel_launch(void* const* d_in, const int* in_sizes, int n_in,
                              void* d_out, int out_size, void* d_ws, size_t ws_size,
                              hipStream_t stream) {
  (void)in_sizes; (void)n_in; (void)out_size; (void)ws_size;
  const float* x   = (const float*)d_in[0];
  const int* eidx  = (const int*)d_in[1];
  const float *gWl[4], *gbl[4], *gWr[4], *gatt[4], *gbias[4];
  for (int i = 0; i < 4; i++) {
    gWl[i]   = (const float*)d_in[2 + i];
    gbl[i]   = (const float*)d_in[6 + i];
    gWr[i]   = (const float*)d_in[10 + i];
    gatt[i]  = (const float*)d_in[14 + i];
    gbias[i] = (const float*)d_in[18 + i];
  }
  const float* Wih[2] = {(const float*)d_in[22], (const float*)d_in[23]};
  const float* Whh[2] = {(const float*)d_in[24], (const float*)d_in[25]};
  const float* bih[2] = {(const float*)d_in[26], (const float*)d_in[27]};
  const float* bhh[2] = {(const float*)d_in[28], (const float*)d_in[29]};
  const float* attnW = (const float*)d_in[30];
  const float* attnB = (const float*)d_in[31];
  const float *clsW[7], *clsB[7];
  for (int i = 0; i < 7; i++) {
    clsW[i] = (const float*)d_in[32 + i];
    clsB[i] = (const float*)d_in[39 + i];
  }
  float* out = (float*)d_out;

  // workspace layout (floats)
  float* ws = (float*)d_ws;
  size_t off = 0;
  auto alloc = [&](size_t n){ float* p = ws + off; off += n; return p; };
  float* H0   = alloc((size_t)BB * NND * CCH);
  float* XL   = alloc((size_t)BB * NND * FEAT);
  float* XR   = alloc((size_t)BB * NND * FEAT);
  float* HB   = alloc((size_t)BB * NND * FEAT);
  float* LG   = alloc((size_t)BB * NE2 * 2);
  float* MB   = alloc((size_t)BB * NND * 2);
  float* DEN  = alloc((size_t)BB * NND * 2);
  float* PRE0 = alloc((size_t)BB * TT * GDIM);
  float* PRE1 = alloc((size_t)BB * TT * GDIM);
  float* LO   = alloc((size_t)BB * TT * 2 * HIDN);
  float* CS   = alloc((size_t)BB * HIDN);
  float* BS   = alloc((size_t)GDIM);
  float* WLG  = alloc((size_t)BB * TT);
  float* LA   = alloc((size_t)BB * 2 * HIDN);
  float* ZB   = alloc((size_t)BB * ((size_t)FEAT * NND + 2 * HIDN));
  float* T1   = alloc((size_t)BB * 512);
  float* T2   = alloc((size_t)BB * 512);
  _Float16* WhhH[2];
  WhhH[0] = (_Float16*)alloc((size_t)GDIM * HIDN / 2);   // 2048x512 f16
  WhhH[1] = (_Float16*)alloc((size_t)GDIM * HIDN / 2);
  _Float16* HSH[2];
  HSH[0] = (_Float16*)alloc((size_t)BB * HIDN / 2);      // 32x512 f16
  HSH[1] = (_Float16*)alloc((size_t)BB * HIDN / 2);
  int* DEG  = (int*)alloc(NND);
  int* ROWP = (int*)alloc(NND + 1);
  int* CSRE = (int*)alloc(NE2);

  auto gemm = [&](const float* A, const float* Bm, const float* bias, float* C,
                  int M, int N, int K, int lda, int ldb, int ldc,
                  long sA, long sB, long sC, int bT, int act, int batch){
    dim3 gr((M + 31) / 32, (N + 63) / 64, batch);
    bool guard = (M % 32) || (N % 64) || (K % 32);
    if (bT) {
      if (guard) gemm_wmma<true, true ><<<gr, 128, 0, stream>>>(A, Bm, bias, C, M, N, K, lda, ldb, ldc, sA, sB, sC, act);
      else       gemm_wmma<true, false><<<gr, 128, 0, stream>>>(A, Bm, bias, C, M, N, K, lda, ldb, ldc, sA, sB, sC, act);
    } else {
      if (guard) gemm_wmma<false, true ><<<gr, 128, 0, stream>>>(A, Bm, bias, C, M, N, K, lda, ldb, ldc, sA, sB, sC, act);
      else       gemm_wmma<false, false><<<gr, 128, 0, stream>>>(A, Bm, bias, C, M, N, K, lda, ldb, ldc, sA, sB, sC, act);
    }
  };

  // ---------- build CSR once (edges shared by all batches & layers) ----------
  k_fill_i<<<g1(NND, 128), 128, 0, stream>>>(DEG, NND, 0);
  k_deg<<<g1(NE2, 256), 256, 0, stream>>>(eidx, DEG);
  k_scan<<<1, 1, 0, stream>>>(DEG, ROWP);
  k_fillcsr<<<1, NND, 0, stream>>>(eidx, ROWP, CSRE);

  // ---------- GAT stack (4 layers, batched over B graphs) ----------
  k_h0<<<g1(BB * NND * CCH, 256), 256, 0, stream>>>(x, H0);
  const float* hin = H0;
  int fi = CCH;
  for (int l = 0; l < 4; l++) {
    gemm(hin, gWl[l], gbl[l], XL, NND, FEAT, fi, fi, FEAT, FEAT,
         (long)NND * fi, 0, (long)NND * FEAT, 0, 0, BB);
    gemm(hin, gWr[l], nullptr, XR, NND, FEAT, fi, fi, FEAT, FEAT,
         (long)NND * fi, 0, (long)NND * FEAT, 0, 0, BB);
    k_edge_logit<<<dim3(NE2 / 4, BB, 2), 128, 0, stream>>>(XL, XR, gatt[l], eidx, LG);
    k_node_maxden<<<dim3(NND, BB), 64, 0, stream>>>(LG, ROWP, CSRE, MB, DEN);
    k_edge_exp<<<g1(BB * NE2 * 2, 256), 256, 0, stream>>>(LG, MB, eidx);
    k_gat_out<<<dim3(NND, BB), 256, 0, stream>>>(LG, DEN, XL, eidx, ROWP, CSRE, gbias[l], HB);
    hin = HB;
    fi = FEAT;
  }

  // ---------- bidirectional LSTM ----------
  k_f2h<<<g1((long)GDIM * HIDN, 256), 256, 0, stream>>>(Whh[0], WhhH[0], (long)GDIM * HIDN);
  k_f2h<<<g1((long)GDIM * HIDN, 256), 256, 0, stream>>>(Whh[1], WhhH[1], (long)GDIM * HIDN);
  float* PRE[2] = {PRE0, PRE1};
  for (int d = 0; d < 2; d++) {
    k_add2<<<g1(GDIM, 256), 256, 0, stream>>>(bih[d], bhh[d], BS, GDIM);
    // pre = xl @ Wih^T + (bih+bhh) ; xl = x reshaped (B*T, C*N)
    gemm(x, Wih[d], BS, PRE[d], BB * TT, GDIM, CCH * NND,
         CCH * NND, CCH * NND, GDIM, 0, 0, 0, 1, 0, 1);
    k_fill<<<g1(BB * HIDN / 2, 256), 256, 0, stream>>>((float*)HSH[0], BB * HIDN / 2, 0.f);
    k_fill<<<g1(BB * HIDN, 256), 256, 0, stream>>>(CS, BB * HIDN, 0.f);
    int p = 0;
    for (int step = 0; step < TT; step++) {
      int t = (d == 0) ? step : (TT - 1 - step);
      k_lstm_step<<<HIDN / 16, 128, 0, stream>>>(HSH[p], WhhH[d], PRE[d], CS,
                                                 HSH[p ^ 1], LO, t, d);
      p ^= 1;
    }
  }

  // ---------- attention pooling ----------
  k_attn_logit<<<dim3(TT / 4, BB), 128, 0, stream>>>(LO, attnW, attnB, WLG);
  k_softmax_T<<<BB, TT, 0, stream>>>(WLG);
  k_attn_pool<<<g1(BB * 2 * HIDN, 256), 256, 0, stream>>>(WLG, LO, LA);

  // ---------- concat z = [xg_flat, la] ----------
  const long ZW = (long)FEAT * NND + 2 * HIDN;  // 132096
  k_build_z<<<g1((long)BB * ZW, 256), 256, 0, stream>>>(HB, LA, ZB);

  // ---------- classifier ----------
  const int dims[8] = {(int)ZW, 512, 256, 256, 256, 64, 32, 12};
  const float* zc = ZB;
  for (int i = 0; i < 7; i++) {
    float* zd = (i == 6) ? out : ((i % 2 == 0) ? T1 : T2);
    gemm(zc, clsW[i], clsB[i], zd, BB, dims[i + 1], dims[i],
         dims[i], dims[i + 1], dims[i + 1], 0, 0, 0, 0, (i < 6) ? 1 : 2, 1);
    zc = zd;
  }
}
